// AGNNConv_26216480375302
// MI455X (gfx1250) — matrix operand order repaired and checked by reading, not compiled, not run on hardware
//
#include <hip/hip_runtime.h>
#include <hip/hip_bf16.h>

typedef __attribute__((ext_vector_type(16))) _Float16 v16h;
typedef __attribute__((ext_vector_type(8)))  _Float16 v8h;
typedef __attribute__((ext_vector_type(8)))  float    v8f;
typedef __attribute__((ext_vector_type(4)))  float    v4f;

#define D_FEAT 128

// Monotone order-preserving float <-> uint encoding so we can do an exact
// segmented max with a single unsigned atomicMax (no CAS loop).
__device__ __forceinline__ unsigned fenc(float f) {
    unsigned u = __float_as_uint(f);
    return (u & 0x80000000u) ? ~u : (u | 0x80000000u);
}
__device__ __forceinline__ float fdec(unsigned u) {
    unsigned v = (u & 0x80000000u) ? (u & 0x7fffffffu) : ~u;
    return __uint_as_float(v);
}

// ---------------------------------------------------------------------------
// K0: initialize seg_max keys (0 == below every encoded float that can occur
// for a node with >=1 incoming edge), seg_sum, and the output accumulator.
// ---------------------------------------------------------------------------
__global__ void agnn_init(unsigned* __restrict__ segmax, float* __restrict__ segsum,
                          float* __restrict__ out, int n_nodes, int out_elems) {
    int i = blockIdx.x * blockDim.x + threadIdx.x;
    int stride = gridDim.x * blockDim.x;
    for (int k = i; k < n_nodes; k += stride) { segmax[k] = 0u; segsum[k] = 0.0f; }
    for (int k = i; k < out_elems; k += stride) out[k] = 0.0f;
}

// ---------------------------------------------------------------------------
// K1: one wave per node. L2-normalize the f32 row (eps matching F.normalize)
// and store as f16 -> halves the gather bytes of the edge-dot phase and feeds
// the f16 WMMA pipe.
// ---------------------------------------------------------------------------
__global__ void agnn_normalize(const float* __restrict__ feat,
                               _Float16* __restrict__ nf, int n_nodes) {
    int wave = (int)((blockIdx.x * blockDim.x + threadIdx.x) >> 5);
    int lane = threadIdx.x & 31;
    if (wave >= n_nodes) return;
    const float* row = feat + (size_t)wave * D_FEAT;
    v4f f = *(const v4f*)(row + lane * 4);
    float ss = f.x * f.x + f.y * f.y + f.z * f.z + f.w * f.w;
    #pragma unroll
    for (int m = 16; m >= 1; m >>= 1) ss += __shfl_xor(ss, m);
    float rn = 1.0f / fmaxf(sqrtf(ss), 1e-12f);
    _Float16* orow = nf + (size_t)wave * D_FEAT + lane * 4;
    orow[0] = (_Float16)(f.x * rn);
    orow[1] = (_Float16)(f.y * rn);
    orow[2] = (_Float16)(f.z * rn);
    orow[3] = (_Float16)(f.w * rn);
}

// ---------------------------------------------------------------------------
// K2: WMMA edge-cosine. One wave per tile of 16 edges.
//   A (16x32 f16, M=edge) : row m = nf[src[tile*16+m]], contiguous K:
//       lanes 0-15 (M=lane)  halves = K kb+0..15
//       lanes16-31 (M=lane-16) halves = K kb+16..31
//   B (32x16 f16, N=edge) : col n = nf[dst[tile*16+n]], contiguous K:
//       lanes 0-15 halves = K kb+0..15 ; lanes 16-31 halves = K kb+16..31
//   4 WMMA steps cover K=128; diag(C) gives the 16 per-edge dots.
// ---------------------------------------------------------------------------
__global__ void agnn_edge_cos_wmma(const _Float16* __restrict__ nf,
                                   const int* __restrict__ src,
                                   const int* __restrict__ dst,
                                   const float* __restrict__ beta,
                                   float* __restrict__ e_ws,
                                   unsigned* __restrict__ segmax,
                                   int ntiles) {
    int wave = (int)((blockIdx.x * blockDim.x + threadIdx.x) >> 5);
    int lane = threadIdx.x & 31;
    if (wave >= ntiles) return;   // uniform per wave: EXEC stays all-ones for WMMA

    int base = wave * 16;
    int m  = lane & 15;
    int hi = lane >> 4;           // which half of the wave

    int sn = src[base + m];
    int dn = dst[base + m];
    const _Float16* arow = nf + (size_t)sn * D_FEAT;
    const _Float16* brow = nf + (size_t)dn * D_FEAT;

    v8f c = {};
    #pragma unroll
    for (int s = 0; s < 4; ++s) {
        int kb = s * 32;
        int offA = kb + hi * 16;
        v8h a0 = *(const v8h*)(arow + offA);
        v8h a1 = *(const v8h*)(arow + offA + 8);
        v16h a;
        #pragma unroll
        for (int i = 0; i < 8; ++i) { a[i] = a0[i]; a[8 + i] = a1[i]; }
        int offB = kb + hi * 16;
        v16h b = *(const v16h*)(brow + offB);
        // (neg_a, A, neg_b, B, c_mod, C, reuse_a, reuse_b)
        c = __builtin_amdgcn_wmma_f32_16x16x32_f16(false, a, false, b,
                                                   (short)0, c, false, false);
    }

    // Diagonal extraction: edge t lives at (lane t, vgpr t) for t<8 and
    // (lane t+16, vgpr t-8) for t>=8. Each diag lane owns exactly one edge,
    // whose dst node id it already loaded (dn).
    int t = (lane < 8) ? lane : ((lane >= 24) ? lane - 16 : -1);
    if (t >= 0) {
        int r = t & 7;
        float v;
        switch (r) {
            case 0: v = c[0]; break; case 1: v = c[1]; break;
            case 2: v = c[2]; break; case 3: v = c[3]; break;
            case 4: v = c[4]; break; case 5: v = c[5]; break;
            case 6: v = c[6]; break; default: v = c[7]; break;
        }
        float e = beta[0] * v;
        e_ws[base + t] = e;
        atomicMax(&segmax[dn], fenc(e));
    }
}

// Scalar tail (E % 16 edges): one wave per edge, VALU dot on f16 rows.
__global__ void agnn_edge_cos_tail(const _Float16* __restrict__ nf,
                                   const int* __restrict__ src,
                                   const int* __restrict__ dst,
                                   const float* __restrict__ beta,
                                   float* __restrict__ e_ws,
                                   unsigned* __restrict__ segmax,
                                   int estart, int E) {
    int wave = (int)((blockIdx.x * blockDim.x + threadIdx.x) >> 5);
    int lane = threadIdx.x & 31;
    int edge = estart + wave;
    if (edge >= E) return;
    const _Float16* a = nf + (size_t)src[edge] * D_FEAT;
    const _Float16* b = nf + (size_t)dst[edge] * D_FEAT;
    float s = 0.0f;
    #pragma unroll
    for (int j = 0; j < 4; ++j)
        s += (float)a[lane * 4 + j] * (float)b[lane * 4 + j];
    #pragma unroll
    for (int mk = 16; mk >= 1; mk >>= 1) s += __shfl_xor(s, mk);
    if (lane == 0) {
        float e = beta[0] * s;
        e_ws[edge] = e;
        atomicMax(&segmax[dst[edge]], fenc(e));
    }
}

// ---------------------------------------------------------------------------
// K3: ex = exp(e - seg_max[dst]); seg_sum[dst] += ex. One thread per edge.
// ---------------------------------------------------------------------------
__global__ void agnn_edge_exp(const float* __restrict__ e_ws,
                              const int* __restrict__ dst,
                              const unsigned* __restrict__ segmax,
                              float* __restrict__ ex_ws,
                              float* __restrict__ segsum, int E) {
    int i = blockIdx.x * blockDim.x + threadIdx.x;
    if (i >= E) return;
    int d = dst[i];
    float ex = expf(e_ws[i] - fdec(segmax[d]));
    ex_ws[i] = ex;
    atomicAdd(&segsum[d], ex);
}

// ---------------------------------------------------------------------------
// K4: out[dst] += (ex/seg_sum[dst]) * feat[src]. One wave per edge; each lane
// owns 4 consecutive features (float4 gather + 4 f32 L2 atomics).
// ---------------------------------------------------------------------------
__global__ void agnn_aggregate(const float* __restrict__ feat,
                               const int* __restrict__ src,
                               const int* __restrict__ dst,
                               const float* __restrict__ ex_ws,
                               const float* __restrict__ segsum,
                               float* __restrict__ out, int E) {
    int wave = (int)((blockIdx.x * blockDim.x + threadIdx.x) >> 5);
    int lane = threadIdx.x & 31;
    if (wave >= E) return;
    int s = src[wave];
    int d = dst[wave];
    float p = ex_ws[wave] / segsum[d];
    v4f f = *(const v4f*)(feat + (size_t)s * D_FEAT + lane * 4);
    float* o = out + (size_t)d * D_FEAT + lane * 4;
    atomicAdd(o + 0, p * f.x);
    atomicAdd(o + 1, p * f.y);
    atomicAdd(o + 2, p * f.z);
    atomicAdd(o + 3, p * f.w);
}

extern "C" void kernel_launch(void* const* d_in, const int* in_sizes, int n_in,
                              void* d_out, int out_size, void* d_ws, size_t ws_size,
                              hipStream_t stream) {
    const float* feat = (const float*)d_in[0];
    const int*   src  = (const int*)d_in[1];
    const int*   dst  = (const int*)d_in[2];
    const float* beta = (const float*)d_in[3];
    float* out = (float*)d_out;

    const int E = in_sizes[1];
    const int N = in_sizes[0] / D_FEAT;

    // Workspace carving (16B aligned): e[E], ex[E], segmax[N], segsum[N], nf[N*128] f16
    char* ws = (char*)d_ws;
    auto align16 = [](size_t x) { return (x + 15) & ~(size_t)15; };
    float*    e_ws   = (float*)ws;            ws += align16((size_t)E * 4);
    float*    ex_ws  = (float*)ws;            ws += align16((size_t)E * 4);
    unsigned* segmax = (unsigned*)ws;         ws += align16((size_t)N * 4);
    float*    segsum = (float*)ws;            ws += align16((size_t)N * 4);
    _Float16* nf     = (_Float16*)ws;

    // K0: init
    agnn_init<<<1024, 256, 0, stream>>>(segmax, segsum, out, N, out_size);

    // K1: normalize -> f16 (one wave per node, 8 waves/block)
    agnn_normalize<<<(N + 7) / 8, 256, 0, stream>>>(feat, nf, N);

    // K2: WMMA edge cosine, 16 edges per wave, 8 waves/block
    int ntiles = E / 16;
    if (ntiles > 0)
        agnn_edge_cos_wmma<<<(ntiles + 7) / 8, 256, 0, stream>>>(
            nf, src, dst, beta, e_ws, segmax, ntiles);
    int tail = E - ntiles * 16;
    if (tail > 0)
        agnn_edge_cos_tail<<<(tail + 7) / 8, 256, 0, stream>>>(
            nf, src, dst, beta, e_ws, segmax, ntiles * 16, E);

    // K3: softmax numerator + segment sum
    agnn_edge_exp<<<(E + 255) / 256, 256, 0, stream>>>(e_ws, dst, segmax, ex_ws, segsum, E);

    // K4: weighted aggregation (one wave per edge, 8 waves/block)
    agnn_aggregate<<<(E + 7) / 8, 256, 0, stream>>>(feat, src, dst, ex_ws, segsum, out, E);
}
